// LinearTemporalInteractionCrossAttention_32942399161112
// MI455X (gfx1250) — matrix operand order repaired and checked by reading, not compiled, not run on hardware
//
#include <hip/hip_runtime.h>
#include <cstdint>

#define BB 32
#define TT 1024
#define DD 512
#define HH 8
#define EE 2048

typedef __attribute__((ext_vector_type(16))) _Float16 v16h;
typedef __attribute__((ext_vector_type(8)))  float    v8f;
typedef unsigned int u32x4 __attribute__((ext_vector_type(4)));
typedef int          i32x8 __attribute__((ext_vector_type(8)));
typedef int          i32x4 __attribute__((ext_vector_type(4)));

union HV { float4 f[2]; v16h h; };

// A fragment (16x32 f16): halves 0..7 at p[0..7], halves 8..15 at p[16..23]
__device__ __forceinline__ v16h load_frag_a(const _Float16* p) {
  HV u;
  u.f[0] = *(const float4*)(p);
  u.f[1] = *(const float4*)(p + 16);
  return u.h;
}
// B fragment (32x16 f16): halves 0..7 at p[0..7], halves 8..15 at p[16..23]
__device__ __forceinline__ v16h load_frag_b(const _Float16* p) {
  HV u;
  u.f[0] = *(const float4*)(p);
  u.f[1] = *(const float4*)(p + 16);
  return u.h;
}

__device__ __forceinline__ float sigmoidf_(float x) { return 1.0f / (1.0f + expf(-x)); }

// ---- CDNA5 async global->LDS copy (ASYNCcnt), 16 bytes per lane -----------
__device__ __forceinline__ void async_b128(uint32_t lds_addr, const void* gptr) {
  asm volatile("global_load_async_to_lds_b128 %0, %1, off"
               :: "v"(lds_addr), "v"(gptr)
               : "memory");
}
__device__ __forceinline__ void wait_async6() {
  asm volatile("s_wait_asynccnt 0x6" ::: "memory");
}
__device__ __forceinline__ void wait_async0() {
  asm volatile("s_wait_asynccnt 0x0" ::: "memory");
}

// ---- CDNA5 TDM: 2D tile load Global->LDS (TENSORcnt) ----------------------
// tile: tile1 rows x tile0 elems (f16), row stride = stride0 elems.
__device__ __forceinline__ void tdm_load_2d_f16(uint32_t lds_addr, uint64_t gaddr,
                                                int tile0, int tile1, int stride0) {
  u32x4 g0;
  g0[0] = 1u;                                     // count=1 (valid descriptor)
  g0[1] = lds_addr;                               // LDS byte address
  g0[2] = (uint32_t)(gaddr & 0xFFFFFFFFu);        // global addr [31:0]
  g0[3] = (uint32_t)((gaddr >> 32) & 0x01FFFFFFu) // global addr [56:32]
          | (2u << 30);                           // type = 2 ("image")
  i32x8 g1;
  g1[0] = (1 << 16);                              // data_size=1 (2 bytes)
  g1[1] = (tile0 & 0xFFFF) << 16;                 // tensor_dim0 [15:0]
  g1[2] = ((tile0 >> 16) & 0xFFFF) | ((tile1 & 0xFFFF) << 16);  // dim0 hi | dim1 lo
  g1[3] = ((tile1 >> 16) & 0xFFFF) | ((tile0 & 0xFFFF) << 16);  // dim1 hi | tile_dim0
  g1[4] = (tile1 & 0xFFFF);                       // tile_dim1 (tile_dim2=0)
  g1[5] = stride0;                                // tensor_dim0_stride [31:0]
  g1[6] = 0;
  g1[7] = 0;
  i32x4 z4 = {0, 0, 0, 0};
  i32x8 z8 = {0, 0, 0, 0, 0, 0, 0, 0};
  __builtin_amdgcn_tensor_load_to_lds(g0, g1, z4, z4, z8, 0);
}

// ---------------------------------------------------------------------------
// Weight transpose + f32 -> f16 convert:  WT[n*K + k] = W[k*N + n]
// ---------------------------------------------------------------------------
__global__ void trW_k(const float* __restrict__ W, _Float16* __restrict__ WT,
                      int K, int N) {
  size_t i = (size_t)blockIdx.x * blockDim.x + threadIdx.x;
  if (i >= (size_t)K * N) return;
  size_t n = i / K, k = i % K;
  WT[i] = (_Float16)W[k * (size_t)N + n];
}

// ---------------------------------------------------------------------------
// LayerNorm of x1 and x2 rows (D=512) -> f16 normalized outputs
// ---------------------------------------------------------------------------
__global__ void __launch_bounds__(256)
ln_k(const float* __restrict__ x1, const float* __restrict__ x2,
     const float* __restrict__ gw, const float* __restrict__ bw,
     _Float16* __restrict__ o1, _Float16* __restrict__ o2) {
  const int row = blockIdx.x;
  const float* x = (blockIdx.y ? x2 : x1) + (size_t)row * DD;
  _Float16*   o = (blockIdx.y ? o2 : o1) + (size_t)row * DD;
  __shared__ float red[256];
  __shared__ float stat[2];
  int t = threadIdx.x;
  float a0 = x[t], a1 = x[t + 256];
  red[t] = a0 + a1;
  __syncthreads();
  for (int s = 128; s > 0; s >>= 1) { if (t < s) red[t] += red[t + s]; __syncthreads(); }
  if (t == 0) stat[0] = red[0] * (1.0f / DD);
  __syncthreads();
  float mean = stat[0];
  float d0 = a0 - mean, d1 = a1 - mean;
  red[t] = d0 * d0 + d1 * d1;
  __syncthreads();
  for (int s = 128; s > 0; s >>= 1) { if (t < s) red[t] += red[t + s]; __syncthreads(); }
  if (t == 0) stat[1] = rsqrtf(red[0] * (1.0f / DD) + 1e-5f);
  __syncthreads();
  float inv = stat[1];
  o[t]       = (_Float16)(d0 * inv * gw[t] + bw[t]);
  o[t + 256] = (_Float16)(d1 * inv * gw[t + 256] + bw[t + 256]);
}

// ---------------------------------------------------------------------------
// Main WMMA GEMM with double-buffered LDS fed by async global->LDS copies.
// (Mtot x K) @ (K x N): A = f16 act (two batch halves), B = f16 WT (N x K).
// Block 256 = 8 waves; tile 128(M) x 256(N); K-step 32.
// MODE 0: +bias -> f16   MODE 1: +bias+mask -> f16   MODE 3: +bias+residual -> f32
// ---------------------------------------------------------------------------
template <int MODE>
__global__ void __launch_bounds__(256)
gemm_k(const _Float16* __restrict__ A0, const _Float16* __restrict__ A1, int MB,
       const _Float16* __restrict__ WT, const float* __restrict__ bias,
       const float* __restrict__ mask,
       const float* __restrict__ res0, const float* __restrict__ res1,
       _Float16* __restrict__ outH, float* __restrict__ outF, int N, int K) {
  __shared__ _Float16 As[2][128 * 32] __attribute__((aligned(16)));
  __shared__ _Float16 Bs[2][256 * 32] __attribute__((aligned(16)));
  const int t = threadIdx.x;
  const int lane = t & 31, wave = t >> 5;
  const int wm = wave >> 2, wn = wave & 3;
  const int r0 = blockIdx.x * 128;
  const int n0 = blockIdx.y * 256;
  const int g = lane >> 4, m = lane & 15;
  const uint32_t as_base = (uint32_t)(uintptr_t)&As[0][0];
  const uint32_t bs_base = (uint32_t)(uintptr_t)&Bs[0][0];

  v8f acc[4][4] = {};

  auto stage = [&](int buf, int k0) {
    {  // A tile: 128 rows x 32 halves; this thread: row t>>1, 32B chunk t&1
      int row = t >> 1, ch = t & 1;
      int r = r0 + row;
      const _Float16* src =
          ((r < MB) ? (A0 + (size_t)r * K) : (A1 + (size_t)(r - MB) * K)) + k0 + ch * 16;
      uint32_t dst = as_base + (uint32_t)buf * (128 * 32 * 2) +
                     (uint32_t)(row * 32 + ch * 16) * 2;
      async_b128(dst, src);
      async_b128(dst + 16, (const char*)src + 16);
    }
    {  // B tile: 256 rows (n) x 32 halves (k); this thread: one 64B row
      const _Float16* src = WT + (size_t)(n0 + t) * K + k0;
      uint32_t dst = bs_base + (uint32_t)buf * (256 * 32 * 2) + (uint32_t)t * 64;
      async_b128(dst, src);
      async_b128(dst + 16, (const char*)src + 16);
      async_b128(dst + 32, (const char*)src + 32);
      async_b128(dst + 48, (const char*)src + 48);
    }
  };

  const int steps = K / 32;
  stage(0, 0);
  for (int kk = 0; kk < steps; ++kk) {
    if (kk + 1 < steps) {
      stage((kk + 1) & 1, (kk + 1) * 32);
      wait_async6();   // async loads complete in order: current tile resident
    } else {
      wait_async0();
    }
    __syncthreads();
    const _Float16* Ac = &As[kk & 1][0];
    const _Float16* Bc = &Bs[kk & 1][0];
    v16h a[4], b[4];
#pragma unroll
    for (int mi = 0; mi < 4; ++mi)
      a[mi] = load_frag_a(Ac + (wm * 64 + mi * 16 + m) * 32 + g * 8);
#pragma unroll
    for (int ni = 0; ni < 4; ++ni)
      b[ni] = load_frag_b(Bc + (wn * 64 + ni * 16 + m) * 32 + g * 8);
#pragma unroll
    for (int mi = 0; mi < 4; ++mi)
#pragma unroll
      for (int ni = 0; ni < 4; ++ni)
        acc[mi][ni] = __builtin_amdgcn_wmma_f32_16x16x32_f16(
            false, a[mi], false, b[ni], (short)0, acc[mi][ni], false, false);
    __syncthreads();
  }
  // epilogue: C layout — M = r + 8*g, N = lane&15
#pragma unroll
  for (int mi = 0; mi < 4; ++mi) {
#pragma unroll
    for (int ni = 0; ni < 4; ++ni) {
#pragma unroll
      for (int r = 0; r < 8; ++r) {
        int row = r0 + wm * 64 + mi * 16 + g * 8 + r;
        int col = n0 + wn * 64 + ni * 16 + m;
        float v = acc[mi][ni][r] + bias[col];
        if constexpr (MODE == 1) v += (1.0f - mask[row]) * (-1000000.0f);
        if constexpr (MODE == 3) {
          float rv = (row < MB) ? res0[(size_t)row * N + col]
                                : res1[(size_t)(row - MB) * N + col];
          outF[(size_t)row * N + col] = v + rv;
        } else {
          outH[(size_t)row * N + col] = (_Float16)v;
        }
      }
    }
  }
}

// ---------------------------------------------------------------------------
// q softmax over last 64 elems (per b,t,h row)
// ---------------------------------------------------------------------------
__global__ void qsm_k(_Float16* __restrict__ q, int nrows) {
  int r = blockIdx.x * blockDim.x + threadIdx.x;
  if (r >= nrows) return;
  _Float16* p = q + (size_t)r * 64;
  float mx = -1e30f;
  for (int i = 0; i < 64; ++i) mx = fmaxf(mx, (float)p[i]);
  float s = 0.0f;
  for (int i = 0; i < 64; ++i) s += expf((float)p[i] - mx);
  float inv = 1.0f / s;
  for (int i = 0; i < 64; ++i) p[i] = (_Float16)(expf((float)p[i] - mx) * inv);
}

// ---------------------------------------------------------------------------
// k softmax over seq axis T per (b2, d) column; coalesced across d
// ---------------------------------------------------------------------------
__global__ void ksm_k(_Float16* __restrict__ kbuf) {
  int b2 = blockIdx.x;
  for (int d = threadIdx.x; d < DD; d += blockDim.x) {
    _Float16* p = kbuf + (size_t)b2 * TT * DD + d;
    float mx = -1e30f;
    for (int tt = 0; tt < TT; ++tt) mx = fmaxf(mx, (float)p[(size_t)tt * DD]);
    float s = 0.0f;
    for (int tt = 0; tt < TT; ++tt) s += expf((float)p[(size_t)tt * DD] - mx);
    float inv = 1.0f / s;
    for (int tt = 0; tt < TT; ++tt)
      p[(size_t)tt * DD] = (_Float16)(expf((float)p[(size_t)tt * DD] - mx) * inv);
  }
}

// ---------------------------------------------------------------------------
// att = k^T v per (b2,h): 64x64, K=T=1024. One wave per block, K/V tiles DMA'd
// into LDS by the Tensor Data Mover (natural [t][dh] layout), fragments
// gathered per-element per the ISA VGPR layout. Output transposed f16 attT[l][d].
// ---------------------------------------------------------------------------
__global__ void __launch_bounds__(32)
att_k(const _Float16* __restrict__ kb, const _Float16* __restrict__ vb,
      _Float16* __restrict__ attT) {
  __shared__ _Float16 Ks2[32 * 64] __attribute__((aligned(16)));  // [t][dh]
  __shared__ _Float16 Vs2[32 * 64] __attribute__((aligned(16)));  // [t][dh]
  const int bh = blockIdx.x;
  const int b2 = bh / HH, h = bh % HH;
  const int ho = h * 64;
  const int lane = threadIdx.x;
  const int g = lane >> 4, m = lane & 15;
  const size_t base = (size_t)b2 * TT * DD + ho;
  const uint32_t ks_lds = (uint32_t)(uintptr_t)&Ks2[0];
  const uint32_t vs_lds = (uint32_t)(uintptr_t)&Vs2[0];
  v8f acc[4][4] = {};
  for (int t0 = 0; t0 < TT; t0 += 32) {
    __syncthreads();
    uint64_t gaK = (uint64_t)(uintptr_t)(kb + base + (size_t)t0 * DD);
    uint64_t gaV = (uint64_t)(uintptr_t)(vb + base + (size_t)t0 * DD);
    tdm_load_2d_f16(ks_lds, gaK, 64, 32, DD);
    tdm_load_2d_f16(vs_lds, gaV, 64, 32, DD);
    __builtin_amdgcn_s_wait_tensorcnt(0);
    __syncthreads();
    v16h a[4], b[4];
#pragma unroll
    for (int mi = 0; mi < 4; ++mi) {
      int mg = mi * 16 + m;
#pragma unroll
      for (int i = 0; i < 16; ++i) {
        int K = (i < 8) ? (i + 8 * g) : (16 + (i - 8) + 8 * g);
        a[mi][i] = Ks2[K * 64 + mg];   // A[m=dh][K=t]
      }
    }
#pragma unroll
    for (int ni = 0; ni < 4; ++ni) {
      int ng = ni * 16 + m;
#pragma unroll
      for (int i = 0; i < 16; ++i)
        b[ni][i] = Vs2[((i < 8) ? (i + 8 * g) : (16 + (i - 8) + 8 * g)) * 64 + ng];  // B[K=t][n=dh]
    }
#pragma unroll
    for (int mi = 0; mi < 4; ++mi)
#pragma unroll
      for (int ni = 0; ni < 4; ++ni)
        acc[mi][ni] = __builtin_amdgcn_wmma_f32_16x16x32_f16(
            false, a[mi], false, b[ni], (short)0, acc[mi][ni], false, false);
  }
#pragma unroll
  for (int mi = 0; mi < 4; ++mi)
#pragma unroll
    for (int ni = 0; ni < 4; ++ni)
#pragma unroll
      for (int r = 0; r < 8; ++r) {
        int dm = mi * 16 + g * 8 + r;   // d index
        int dn = ni * 16 + m;           // l index
        attT[((size_t)bh * 64 + dn) * 64 + dm] = (_Float16)acc[mi][ni][r];
      }
}

// ---------------------------------------------------------------------------
// y = q @ att per (b2,h): (T x 64) @ (64 x 64). Block 256 = 8 waves, each wave
// 64 rows (M_TILE=512). B fragments read directly from global attT (K-contig).
// ---------------------------------------------------------------------------
__global__ void __launch_bounds__(256)
ygemm_k(const _Float16* __restrict__ q, const _Float16* __restrict__ attT,
        float* __restrict__ y) {
  __shared__ _Float16 As[512 * 32] __attribute__((aligned(16)));
  const int t = threadIdx.x;
  const int lane = t & 31, wave = t >> 5;
  const int g = lane >> 4, m = lane & 15;
  const int h = blockIdx.y, ho = h * 64;
  const int r0 = blockIdx.x * 512;
  const int b2 = r0 / TT;  // 512 | T so block stays in one batch
  const size_t bh64 = ((size_t)(b2 * HH + h)) * 64;
  v8f acc[4][4] = {};
  for (int k0 = 0; k0 < 64; k0 += 32) {
    __syncthreads();
#pragma unroll
    for (int rr = 0; rr < 2; ++rr) {
      int row = t * 2 + rr;
      const _Float16* src = q + ((size_t)(r0 + row)) * DD + ho + k0;
      float4 v0 = ((const float4*)src)[0];
      float4 v1 = ((const float4*)src)[1];
      float4* dst = (float4*)(As + row * 32);
      dst[0] = v0; dst[1] = v1;
    }
    __syncthreads();
    v16h a[4], b[4];
#pragma unroll
    for (int mi = 0; mi < 4; ++mi)
      a[mi] = load_frag_a(As + (wave * 64 + mi * 16 + m) * 32 + g * 8);
#pragma unroll
    for (int ni = 0; ni < 4; ++ni)
      b[ni] = load_frag_b(attT + (bh64 + ni * 16 + m) * 64 + k0 + g * 8);
#pragma unroll
    for (int mi = 0; mi < 4; ++mi)
#pragma unroll
      for (int ni = 0; ni < 4; ++ni)
        acc[mi][ni] = __builtin_amdgcn_wmma_f32_16x16x32_f16(
            false, a[mi], false, b[ni], (short)0, acc[mi][ni], false, false);
  }
#pragma unroll
  for (int mi = 0; mi < 4; ++mi)
#pragma unroll
    for (int ni = 0; ni < 4; ++ni)
#pragma unroll
      for (int r = 0; r < 8; ++r) {
        int row = r0 + wave * 64 + mi * 16 + g * 8 + r;
        int col = ho + ni * 16 + m;
        y[(size_t)row * DD + col] = acc[mi][ni][r];
      }
}

// ---------------------------------------------------------------------------
// silu(emb) elementwise
// ---------------------------------------------------------------------------
__global__ void silu_k(const float* __restrict__ x, float* __restrict__ o, int n) {
  int i = blockIdx.x * blockDim.x + threadIdx.x;
  if (i < n) { float v = x[i]; o[i] = v * sigmoidf_(v); }
}

// ---------------------------------------------------------------------------
// eo = silu(emb) @ W_emb + b_emb  (64 x 2048) @ (2048 x 1024)
// ---------------------------------------------------------------------------
__global__ void eo_k(const float* __restrict__ sil, const float* __restrict__ Wemb,
                     const float* __restrict__ bemb, float* __restrict__ eo) {
  int b2 = blockIdx.x;
  const float* sr = sil + (size_t)b2 * EE;
  for (int n = threadIdx.x; n < 2 * DD; n += blockDim.x) {
    float acc = bemb[n];
    for (int e = 0; e < EE; ++e) acc += sr[e] * Wemb[(size_t)e * (2 * DD) + n];
    eo[(size_t)b2 * (2 * DD) + n] = acc;
  }
}

// ---------------------------------------------------------------------------
// s = silu(LN(y)*(1+scale)+shift) -> f16
// ---------------------------------------------------------------------------
__global__ void __launch_bounds__(256)
styl_k(const float* __restrict__ y, const float* __restrict__ gw,
       const float* __restrict__ bw, const float* __restrict__ eo,
       _Float16* __restrict__ s) {
  const int row = blockIdx.x;
  const int b2 = row / TT;
  const float* yr = y + (size_t)row * DD;
  const float* er = eo + (size_t)b2 * (2 * DD);
  __shared__ float red[256];
  __shared__ float stat[2];
  int t = threadIdx.x;
  float a0 = yr[t], a1 = yr[t + 256];
  red[t] = a0 + a1;
  __syncthreads();
  for (int q = 128; q > 0; q >>= 1) { if (t < q) red[t] += red[t + q]; __syncthreads(); }
  if (t == 0) stat[0] = red[0] * (1.0f / DD);
  __syncthreads();
  float mean = stat[0];
  float d0 = a0 - mean, d1 = a1 - mean;
  red[t] = d0 * d0 + d1 * d1;
  __syncthreads();
  for (int q = 128; q > 0; q >>= 1) { if (t < q) red[t] += red[t + q]; __syncthreads(); }
  if (t == 0) stat[1] = rsqrtf(red[0] * (1.0f / DD) + 1e-5f);
  __syncthreads();
  float inv = stat[1];
  float v0 = d0 * inv * gw[t] + bw[t];
  v0 = v0 * (1.0f + er[t]) + er[DD + t];
  s[(size_t)row * DD + t] = (_Float16)(v0 * sigmoidf_(v0));
  float v1 = d1 * inv * gw[t + 256] + bw[t + 256];
  v1 = v1 * (1.0f + er[t + 256]) + er[DD + t + 256];
  s[(size_t)row * DD + t + 256] = (_Float16)(v1 * sigmoidf_(v1));
}

// ---------------------------------------------------------------------------
extern "C" void kernel_launch(void* const* d_in, const int* in_sizes, int n_in,
                              void* d_out, int out_size, void* d_ws, size_t ws_size,
                              hipStream_t stream) {
  (void)in_sizes; (void)n_in; (void)out_size; (void)ws_size;
  const float* x1   = (const float*)d_in[0];
  const float* x2   = (const float*)d_in[1];
  const float* emb  = (const float*)d_in[2];
  const float* mask = (const float*)d_in[3];
  // d_in[4] timesteps: unused by reference
  const float* ln_g = (const float*)d_in[5];
  const float* ln_b = (const float*)d_in[6];
  const float* Wq   = (const float*)d_in[7];
  const float* bq   = (const float*)d_in[8];
  const float* Wk   = (const float*)d_in[9];
  const float* bk   = (const float*)d_in[10];
  const float* Wv   = (const float*)d_in[11];
  const float* bv   = (const float*)d_in[12];
  const float* sb_g = (const float*)d_in[13];
  const float* sb_b = (const float*)d_in[14];
  const float* Wemb = (const float*)d_in[15];
  const float* bemb = (const float*)d_in[16];
  const float* Wout = (const float*)d_in[17];
  const float* bout = (const float*)d_in[18];
  float* out = (float*)d_out;

  constexpr size_t MB = (size_t)BB * TT;  // 32768 rows per batch-half
  constexpr size_t MT = 2 * MB;           // 65536 total rows

  char* ws = (char*)d_ws;
  size_t off = 0;
  auto alloc = [&](size_t bytes) {
    char* p = ws + off;
    off = (off + bytes + 255) & ~(size_t)255;
    return p;
  };
  _Float16* hx1n = (_Float16*)alloc(MB * DD * 2);
  _Float16* hx2n = (_Float16*)alloc(MB * DD * 2);
  _Float16* qlin = (_Float16*)alloc(MT * DD * 2);
  _Float16* klin = (_Float16*)alloc(MT * DD * 2);
  _Float16* vlin = (_Float16*)alloc(MT * DD * 2);
  _Float16* WqT  = (_Float16*)alloc((size_t)DD * DD * 2);
  _Float16* WkT  = (_Float16*)alloc((size_t)DD * DD * 2);
  _Float16* WvT  = (_Float16*)alloc((size_t)DD * DD * 2);
  _Float16* WoT  = (_Float16*)alloc((size_t)DD * DD * 2);
  _Float16* attT = (_Float16*)alloc((size_t)2 * BB * HH * 64 * 64 * 2);
  float*    ybuf = (float*)alloc(MT * DD * 4);
  _Float16* sbuf = (_Float16*)alloc(MT * DD * 2);
  float*    silb = (float*)alloc((size_t)2 * BB * EE * 4);
  float*    eob  = (float*)alloc((size_t)2 * BB * 2 * DD * 4);

  const int thr = 256;
  trW_k<<<(DD * DD + thr - 1) / thr, thr, 0, stream>>>(Wq, WqT, DD, DD);
  trW_k<<<(DD * DD + thr - 1) / thr, thr, 0, stream>>>(Wk, WkT, DD, DD);
  trW_k<<<(DD * DD + thr - 1) / thr, thr, 0, stream>>>(Wv, WvT, DD, DD);
  trW_k<<<(DD * DD + thr - 1) / thr, thr, 0, stream>>>(Wout, WoT, DD, DD);

  ln_k<<<dim3((unsigned)MB, 2), 256, 0, stream>>>(x1, x2, ln_g, ln_b, hx1n, hx2n);

  dim3 gg((unsigned)(MT / 128), (unsigned)(DD / 256));
  // n1c = [LN(x1); LN(x2)], n2c = [LN(x2); LN(x1)]
  gemm_k<0><<<gg, 256, 0, stream>>>(hx1n, hx2n, (int)MB, WqT, bq,
                                    nullptr, nullptr, nullptr, qlin, nullptr, DD, DD);
  gemm_k<1><<<gg, 256, 0, stream>>>(hx2n, hx1n, (int)MB, WkT, bk,
                                    mask, nullptr, nullptr, klin, nullptr, DD, DD);
  gemm_k<0><<<gg, 256, 0, stream>>>(hx2n, hx1n, (int)MB, WvT, bv,
                                    nullptr, nullptr, nullptr, vlin, nullptr, DD, DD);

  qsm_k<<<(int)((MT * HH + 255) / 256), 256, 0, stream>>>(qlin, (int)(MT * HH));
  ksm_k<<<2 * BB, 256, 0, stream>>>(klin);

  att_k<<<2 * BB * HH, 32, 0, stream>>>(klin, vlin, attT);
  ygemm_k<<<dim3((unsigned)(MT / 512), HH), 256, 0, stream>>>(qlin, attT, ybuf);

  silu_k<<<(2 * BB * EE + 255) / 256, 256, 0, stream>>>(emb, silb, 2 * BB * EE);
  eo_k<<<2 * BB, 256, 0, stream>>>(silb, Wemb, bemb, eob);

  styl_k<<<(unsigned)MT, 256, 0, stream>>>(ybuf, sb_g, sb_b, eob, sbuf);

  // out = x1c + silu(...) @ W_out + b_out  (f32, written to d_out)
  gemm_k<3><<<gg, 256, 0, stream>>>(sbuf, sbuf + MB * DD, (int)MB, WoT, bout,
                                    nullptr, x1, x2, nullptr, out, DD, DD);
}
